// SNNDense2sparse_20538533609759
// MI455X (gfx1250) — compile-verified, matrix operand order checked by reading, not run on hardware
//
#include <hip/hip_runtime.h>

// ---------------------------------------------------------------------------
// SNN dense->sparse for MI455X (gfx1250, wave32)
//   x:[256,100,700] f32, w:[700,1024] f32
//   out = concat( spk:[256,100,1024] f32 , idx:[6553600] i32 (as raw bits) )
// Phase 0: pack w -> bf16, WMMA-B-fragment-native layout (zeros for K=700..703)
// Phase 1: bf16 WMMA GEMM h = x@w  (M=25600, K=700, N=1024), h -> spk region
// Phase 2: LIF scan over T, in place (h -> spikes)
// Phase 3: ordered nonzero compaction (count / scan / scatter), -1 padding
// ---------------------------------------------------------------------------

typedef __attribute__((ext_vector_type(16))) __bf16        v16bf;
typedef __attribute__((ext_vector_type(8)))  float         v8f;
typedef __attribute__((ext_vector_type(8)))  unsigned int  v8u;

#define SNN_B    256
#define SNN_T    100
#define SNN_IN   700
#define SNN_OUT  1024
#define SNN_M    (SNN_B * SNN_T)          // 25600
#define SNN_NSPK (SNN_M * SNN_OUT)        // 26214400
#define SNN_NIDX (SNN_NSPK / 4)           // 6553600
#define SNN_KT   22                       // ceil(700/32) K-tiles

#define CMP_PER_THREAD 16
#define CMP_BLOCK      256
#define CMP_CHUNK      (CMP_BLOCK * CMP_PER_THREAD)   // 4096
#define CMP_NBLOCKS    (SNN_NSPK / CMP_CHUNK)         // 6400

// d_ws layout
#define WS_COUNTS_OFF   0                          // 6400 ints
#define WS_OFFSETS_OFF  (CMP_NBLOCKS)              // 6400 ints
#define WS_WPACK_BYTES  (51200)                    // = 2*6400*4, 32B aligned
// wPack: [SNN_KT][64 ntiles][32 lanes][16 halfs]  = 22*64*32*32 B = 1.44 MB

static __device__ __forceinline__ unsigned int pack_bf16_pair(float lo, float hi) {
  unsigned int r;
  asm("v_cvt_pk_bf16_f32 %0, %1, %2" : "=v"(r) : "v"(lo), "v"(hi));
  return r;
}

// ---------------------------------------------------------------------------
// Phase 0: pack W (f32 [700,1024]) into bf16 B-fragment layout.
// One thread per packed uint (pair of adjacent-K bf16 values).
// tid = ((kt*64 + nt)*32 + lane)*8 + jp
// lane<16  holds col nt*16+lane,      K = kt*32 +  {0..7, 16..23}
// lane>=16 holds col nt*16+(lane-16), K = kt*32 + {8..15, 24..31}
// K >= 700 slots are zero-filled, so the GEMM B path needs no guards.
// ---------------------------------------------------------------------------
__global__ __launch_bounds__(256) void snn_pack_w(const float* __restrict__ W,
                                                  unsigned int* __restrict__ wPackU) {
  const int tid  = blockIdx.x * blockDim.x + threadIdx.x;  // 0 .. 360447
  const int jp   = tid & 7;
  const int lane = (tid >> 3) & 31;
  const int nt   = (tid >> 8) & 63;
  const int kt   = tid >> 14;                               // 0..21
  const int j    = jp * 2;                                  // 0,2,..,14
  const int col  = nt * 16 + (lane & 15);
  const int k0   = kt * 32 + (lane >> 4) * 8 + ((j < 8) ? j : j + 8);
  const int k1   = k0 + 1;
  const float f0 = (k0 < SNN_IN) ? W[k0 * SNN_OUT + col] : 0.0f;
  const float f1 = (k1 < SNN_IN) ? W[k1 * SNN_OUT + col] : 0.0f;
  const __bf16 b0 = (__bf16)f0;
  const __bf16 b1 = (__bf16)f1;
  unsigned int u = (unsigned int)__builtin_bit_cast(unsigned short, b0) |
                   ((unsigned int)__builtin_bit_cast(unsigned short, b1) << 16);
  wPackU[tid] = u;
}

// ---------------------------------------------------------------------------
// Phase 1: GEMM. Block tile 128(M) x 128(N); 8 waves = 4(M) x 2(N);
// each wave owns 32x64 = 2x4 WMMA 16x16 tiles. K stepped by 32 (bf16 WMMA K).
// A: float4 loads from X (rows are 16B aligned: 700*4 = 2800 = 16*175),
//    packed pairwise with v_cvt_pk_bf16_f32.
// B: two b128 loads from fragment-native wPack (fully coalesced).
// ---------------------------------------------------------------------------
__global__ __launch_bounds__(256) void snn_gemm_bf16(const float* __restrict__ X,
                                                     const v16bf* __restrict__ wPack,
                                                     float* __restrict__ H) {
  const int lane    = threadIdx.x & 31;
  const int wave    = threadIdx.x >> 5;      // 0..7
  const int wm      = wave & 3;              // 4 waves along M
  const int wn      = wave >> 2;             // 2 waves along N
  const int mBlock  = (blockIdx.x % (SNN_M / 128)) * 128;
  const int nBlock  = (blockIdx.x / (SNN_M / 128)) * 128;
  const int mWave   = mBlock + wm * 32;
  const int ntWave  = nBlock / 16 + wn * 4;  // first of this wave's 4 N-tiles
  const int halfSel = lane >> 4;             // 0: lanes 0-15, 1: lanes 16-31
  const int l       = lane & 15;

  v8f acc[2][4];
#pragma unroll
  for (int i = 0; i < 2; ++i)
#pragma unroll
    for (int j = 0; j < 4; ++j) acc[i][j] = (v8f)0.0f;

  // ---- main K loop: 21 full tiles (K = 0..671), unguarded ----
  for (int kt = 0; kt < SNN_KT - 1; ++kt) {
    const int kLo = kt * 32 + halfSel * 8;
    v16bf a[2], b[4];
#pragma unroll
    for (int i = 0; i < 2; ++i) {
      const int row = mWave + i * 16 + l;
      const float4* ap4 = (const float4*)(X + row * SNN_IN + kLo);  // 16B aligned
      const float4 q0 = ap4[0];      // K = kLo+0..3
      const float4 q1 = ap4[1];      // K = kLo+4..7
      const float4 q2 = ap4[4];      // K = kLo+16..19
      const float4 q3 = ap4[5];      // K = kLo+20..23
      v8u up;
      up[0] = pack_bf16_pair(q0.x, q0.y);
      up[1] = pack_bf16_pair(q0.z, q0.w);
      up[2] = pack_bf16_pair(q1.x, q1.y);
      up[3] = pack_bf16_pair(q1.z, q1.w);
      up[4] = pack_bf16_pair(q2.x, q2.y);
      up[5] = pack_bf16_pair(q2.z, q2.w);
      up[6] = pack_bf16_pair(q3.x, q3.y);
      up[7] = pack_bf16_pair(q3.z, q3.w);
      a[i] = __builtin_bit_cast(v16bf, up);
    }
#pragma unroll
    for (int i = 0; i < 4; ++i)
      b[i] = wPack[(kt * 64 + (ntWave + i)) * 32 + lane];   // 2 x b128, coalesced
#pragma unroll
    for (int i = 0; i < 2; ++i)
#pragma unroll
      for (int j = 0; j < 4; ++j)
        acc[i][j] = __builtin_amdgcn_wmma_f32_16x16x32_bf16(
            false, a[i], false, b[j], (short)0, acc[i][j], false, false);
  }

  // ---- K remainder tile kt=21: A guarded (K 672..699); B already zero-padded ----
  {
    const int kt  = SNN_KT - 1;
    const int kLo = kt * 32 + halfSel * 8;
    v16bf a[2], b[4];
#pragma unroll
    for (int i = 0; i < 2; ++i) {
      const int row = mWave + i * 16 + l;
#pragma unroll
      for (int j = 0; j < 8; ++j) {
        const int kA = kLo + j;
        const int kB = kA + 16;
        const int cA = (kA < SNN_IN) ? kA : (SNN_IN - 1);   // clamp: no OOB load
        const int cB = (kB < SNN_IN) ? kB : (SNN_IN - 1);
        const float va = X[row * SNN_IN + cA];
        const float vb = X[row * SNN_IN + cB];
        a[i][j]     = (kA < SNN_IN) ? (__bf16)va : (__bf16)0.0f;
        a[i][8 + j] = (kB < SNN_IN) ? (__bf16)vb : (__bf16)0.0f;
      }
    }
#pragma unroll
    for (int i = 0; i < 4; ++i)
      b[i] = wPack[(kt * 64 + (ntWave + i)) * 32 + lane];
#pragma unroll
    for (int i = 0; i < 2; ++i)
#pragma unroll
      for (int j = 0; j < 4; ++j)
        acc[i][j] = __builtin_amdgcn_wmma_f32_16x16x32_bf16(
            false, a[i], false, b[j], (short)0, acc[i][j], false, false);
  }

  // ---- store C: VGPR r, lanes 0-15 -> (M=r, N=l); lanes 16-31 -> (M=r+8) ----
#pragma unroll
  for (int i = 0; i < 2; ++i) {
#pragma unroll
    for (int j = 0; j < 4; ++j) {
      const int col = (ntWave + j) * 16 + l;
#pragma unroll
      for (int r = 0; r < 8; ++r) {
        const int row = mWave + i * 16 + halfSel * 8 + r;
        H[row * SNN_OUT + col] = acc[i][j][r];
      }
    }
  }
}

// ---------------------------------------------------------------------------
// Phase 2: LIF scan along T, in place. One thread per (b, o) neuron.
// ---------------------------------------------------------------------------
__global__ __launch_bounds__(256) void snn_lif(float* __restrict__ H) {
  const int tid = blockIdx.x * blockDim.x + threadIdx.x;   // 0 .. B*OUT-1
  const int b = tid >> 10;
  const int o = tid & (SNN_OUT - 1);
  float syn = 0.0f, mem = 0.0f;
  int addr = (b * SNN_T) * SNN_OUT + o;
  for (int t = 0; t < SNN_T; ++t) {
    const float h = H[addr];
    syn = 0.9f * syn + h;
    mem = 0.85f * mem + syn;
    const float spk = (mem - 1.0f > 0.0f) ? 1.0f : 0.0f;
    mem -= spk;                      // reset by subtraction (THRESH = 1)
    H[addr] = spk;
    addr += SNN_OUT;
  }
}

// ---------------------------------------------------------------------------
// Phase 3: ordered compaction. fill(-1) -> count -> scan -> scatter.
// ---------------------------------------------------------------------------
__global__ void snn_fill_i32(int* __restrict__ p, int n, int val) {
  const int i = blockIdx.x * blockDim.x + threadIdx.x;
  if (i < n) p[i] = val;
}

__global__ __launch_bounds__(CMP_BLOCK) void snn_count(const float* __restrict__ spk,
                                                       int* __restrict__ blockCounts) {
  __shared__ int sdata[CMP_BLOCK];
  const int base = blockIdx.x * CMP_CHUNK + threadIdx.x * CMP_PER_THREAD;
  int c = 0;
#pragma unroll
  for (int j = 0; j < CMP_PER_THREAD; ++j) c += (spk[base + j] > 0.0f) ? 1 : 0;
  sdata[threadIdx.x] = c;
  __syncthreads();
  for (int s = CMP_BLOCK / 2; s > 0; s >>= 1) {
    if (threadIdx.x < s) sdata[threadIdx.x] += sdata[threadIdx.x + s];
    __syncthreads();
  }
  if (threadIdx.x == 0) blockCounts[blockIdx.x] = sdata[0];
}

__global__ __launch_bounds__(256) void snn_scan_offsets(const int* __restrict__ counts,
                                                        int* __restrict__ offsets,
                                                        int nblocks) {
  __shared__ int sdata[256];
  __shared__ int carry;
  if (threadIdx.x == 0) carry = 0;
  __syncthreads();
  for (int base = 0; base < nblocks; base += 256) {
    const int i = base + threadIdx.x;
    const int v = (i < nblocks) ? counts[i] : 0;
    sdata[threadIdx.x] = v;
    __syncthreads();
    for (int s = 1; s < 256; s <<= 1) {               // Hillis-Steele inclusive
      const int t = (threadIdx.x >= s) ? sdata[threadIdx.x - s] : 0;
      __syncthreads();
      sdata[threadIdx.x] += t;
      __syncthreads();
    }
    const int incl = sdata[threadIdx.x];
    if (i < nblocks) offsets[i] = carry + (incl - v); // exclusive + carry
    __syncthreads();
    if (threadIdx.x == 255) carry += incl;            // incl of lane 255 = total
    __syncthreads();
  }
}

__global__ __launch_bounds__(CMP_BLOCK) void snn_scatter(const float* __restrict__ spk,
                                                         const int* __restrict__ blockOffsets,
                                                         int* __restrict__ idxOut,
                                                         int maxIdx) {
  __shared__ int sdata[CMP_BLOCK];
  const int base = blockIdx.x * CMP_CHUNK + threadIdx.x * CMP_PER_THREAD;
  int c = 0;
#pragma unroll
  for (int j = 0; j < CMP_PER_THREAD; ++j) c += (spk[base + j] > 0.0f) ? 1 : 0;
  sdata[threadIdx.x] = c;
  __syncthreads();
  for (int s = 1; s < CMP_BLOCK; s <<= 1) {           // inclusive scan of counts
    const int t = (threadIdx.x >= s) ? sdata[threadIdx.x - s] : 0;
    __syncthreads();
    sdata[threadIdx.x] += t;
    __syncthreads();
  }
  int pos = blockOffsets[blockIdx.x] + (sdata[threadIdx.x] - c);  // stable rank
  for (int j = 0; j < CMP_PER_THREAD; ++j) {
    if (spk[base + j] > 0.0f) {
      if (pos < maxIdx) idxOut[pos] = base + j;
      ++pos;
    }
  }
}

// ---------------------------------------------------------------------------
extern "C" void kernel_launch(void* const* d_in, const int* in_sizes, int n_in,
                              void* d_out, int out_size, void* d_ws, size_t ws_size,
                              hipStream_t stream) {
  const float* X = (const float*)d_in[0];   // [256,100,700]
  const float* W = (const float*)d_in[1];   // [700,1024]
  float* spk  = (float*)d_out;                          // [25600,1024]
  int*   idxO = (int*)((float*)d_out + SNN_NSPK);       // [6553600]
  int*   blockCounts  = (int*)d_ws + WS_COUNTS_OFF;     // 6400 ints
  int*   blockOffsets = (int*)d_ws + WS_OFFSETS_OFF;    // 6400 ints
  unsigned int* wPackU = (unsigned int*)((char*)d_ws + WS_WPACK_BYTES);
  const v16bf*  wPack  = (const v16bf*)wPackU;          // 1.44 MB, 32B aligned

  // Phase 0: pack W into bf16 fragment-native layout (zero-padded K tail)
  snn_pack_w<<<SNN_KT * 64, 256, 0, stream>>>(W, wPackU);
  // Phase 1: GEMM (h written into spk region)
  snn_gemm_bf16<<<(SNN_M / 128) * (SNN_OUT / 128), 256, 0, stream>>>(X, wPack, spk);
  // Phase 2: LIF recurrence, in place
  snn_lif<<<(SNN_B * SNN_OUT) / 256, 256, 0, stream>>>(spk);
  // Phase 3: ordered nonzero with -1 padding
  snn_fill_i32<<<(SNN_NIDX + 255) / 256, 256, 0, stream>>>(idxO, SNN_NIDX, -1);
  snn_count<<<CMP_NBLOCKS, CMP_BLOCK, 0, stream>>>(spk, blockCounts);
  snn_scan_offsets<<<1, 256, 0, stream>>>(blockCounts, blockOffsets, CMP_NBLOCKS);
  snn_scatter<<<CMP_NBLOCKS, CMP_BLOCK, 0, stream>>>(spk, blockOffsets, idxO, SNN_NIDX);
}
